// MultiHeadAttentionExplain_63428077027529
// MI455X (gfx1250) — compile-verified, matrix operand order checked by reading, not run on hardware
//
#include <hip/hip_runtime.h>

typedef __attribute__((ext_vector_type(16))) _Float16 v16h;
typedef __attribute__((ext_vector_type(8)))  _Float16 v8h;
typedef __attribute__((ext_vector_type(8)))  float    v8f;
typedef __attribute__((ext_vector_type(4)))  int      v4i;

#define GAS __attribute__((address_space(1)))
#define LAS __attribute__((address_space(3)))

#define D_MODEL 1024
#define N_HEADS 16
#define D_HEAD  64
#define BATCH   2
#define SEQ     2048
#define NTOK    (BATCH * SEQ)  // 4096

#if defined(__HIP_DEVICE_COMPILE__) && defined(__has_builtin)
#if __has_builtin(__builtin_amdgcn_global_load_async_to_lds_b128) && \
    __has_builtin(__builtin_amdgcn_s_wait_asynccnt)
#define HAVE_ASYNC_LDS 1
#endif
#endif
#ifndef HAVE_ASYNC_LDS
#define HAVE_ASYNC_LDS 0
#endif

static __device__ __forceinline__ v8f wmma16x16x32(v16h a, v16h b, v8f c) {
  // D = A(16x32 f16) * B(32x16 f16) + C(16x16 f32)
  return __builtin_amdgcn_wmma_f32_16x16x32_f16(false, a, false, b, (short)0, c,
                                                false, false);
}

// A/B fragment for v_wmma_f32_16x16x32_f16 (16-bit operand layout, wave32):
// lane holds row/col = lane&15; halfLane = lane>>4 selects K base 8*hl.
// p points at row_base + k0 + 8*hl; elements 0..7 <- p[0..7] (one b128),
// elements 8..15 <- p[16..23] (one b128).   K = 16*group + 8*hl + j.
static __device__ __forceinline__ v16h frag16(const _Float16* __restrict__ p) {
  const v8h c0 = *(const v8h*)p;
  const v8h c1 = *(const v8h*)(p + 16);
  return __builtin_shufflevector(c0, c1, 0, 1, 2, 3, 4, 5, 6, 7,
                                 8, 9, 10, 11, 12, 13, 14, 15);
}

// ---------------------------------------------------------------------------
// Kernel 0: elementwise f32 -> f16 (8 elements per thread, b128 in/out)
// ---------------------------------------------------------------------------
__global__ __launch_bounds__(256)
void cvt_f16_kernel(const float* __restrict__ in, _Float16* __restrict__ out,
                    int n8) {
  const int i = blockIdx.x * 256 + threadIdx.x;
  if (i >= n8) return;
  const float4 a = ((const float4*)in)[i * 2 + 0];
  const float4 b = ((const float4*)in)[i * 2 + 1];
  union { float4 v; _Float16 h[8]; } u;
  u.h[0] = (_Float16)a.x; u.h[1] = (_Float16)a.y;
  u.h[2] = (_Float16)a.z; u.h[3] = (_Float16)a.w;
  u.h[4] = (_Float16)b.x; u.h[5] = (_Float16)b.y;
  u.h[6] = (_Float16)b.z; u.h[7] = (_Float16)b.w;
  ((float4*)out)[i] = u.v;
}

// ---------------------------------------------------------------------------
// Kernel 1: Y[b,h,l,d] (f16) = X(f16) @ W(f16)^T + bias   (head-split out)
// One wave -> 32(M) x 64(N) tile, 8 accumulators, pure-f16 hot loop.
// ---------------------------------------------------------------------------
__global__ __launch_bounds__(256)
void qkv_proj_kernel(const _Float16* __restrict__ X,
                     const _Float16* __restrict__ W,
                     const float* __restrict__ bias,
                     _Float16* __restrict__ Y) {
  const int lane = threadIdx.x & 31;
  const int wave = threadIdx.x >> 5;
  const int w    = blockIdx.x * 8 + wave;  // 0..2047
  const int mtile = w >> 4;                // 0..127  (32 tokens each)
  const int n64   = w & 15;                // 0..15   (64 out-channels each)
  const int hl = lane >> 4, lm = lane & 15;

  const _Float16* x0 = X + (size_t)(mtile * 32 + lm) * D_MODEL + 8 * hl;
  const _Float16* x1 = x0 + (size_t)16 * D_MODEL;
  const _Float16* wb = W + (size_t)(n64 * 64 + lm) * D_MODEL + 8 * hl;

  v8f acc[2][4] = {};
#pragma unroll 2
  for (int k0 = 0; k0 < D_MODEL; k0 += 32) {
    const v16h a0 = frag16(x0 + k0);
    const v16h a1 = frag16(x1 + k0);
    __builtin_prefetch(wb + k0 + 32, 0, 1);
#pragma unroll
    for (int t = 0; t < 4; ++t) {
      const v16h bf = frag16(wb + (size_t)t * 16 * D_MODEL + k0);
      acc[0][t] = wmma16x16x32(a0, bf, acc[0][t]);
      acc[1][t] = wmma16x16x32(a1, bf, acc[1][t]);
    }
  }
  // C/D layout: lane,reg r -> m = r + 8*hl, n = lane&15
#pragma unroll
  for (int t = 0; t < 4; ++t) {
    const int ncol = n64 * 64 + t * 16 + lm;
    const float bb = bias[ncol];
    const int h = ncol >> 6, d = ncol & 63;
#pragma unroll
    for (int u = 0; u < 2; ++u)
#pragma unroll
      for (int r = 0; r < 8; ++r) {
        const int tok = mtile * 32 + u * 16 + r + 8 * hl;
        const int b = tok >> 11, l = tok & (SEQ - 1);
        Y[(((size_t)b * N_HEADS + h) * SEQ + l) * D_HEAD + d] =
            (_Float16)(acc[u][t][r] + bb);
      }
  }
}

// ---------------------------------------------------------------------------
// Kernel 2: causal flash attention, one wave per (b,h, 32-row q tile).
// Q,K,V in [B,H,L,64] f16. Output merged heads [B,L,1024] f16.
// All 32 lanes own one softmax row; V staged via async LDS loads if available.
// ---------------------------------------------------------------------------
__global__ __launch_bounds__(32)
void flash_attn_kernel(const _Float16* __restrict__ Qh,
                       const _Float16* __restrict__ Kh,
                       const _Float16* __restrict__ Vh,
                       _Float16* __restrict__ Hm) {
  __shared__ float    sS[32][32];   // raw scores tile
  __shared__ _Float16 sP[32][32];   // exp(scores) tile (A operand of PV)
  __shared__ _Float16 sV[32][64];   // staged V block
  __shared__ float    sScale[32];
  __shared__ float    sSum[32];

  const int lane = threadIdx.x;
  const int hl = lane >> 4, lm = lane & 15;
  const int qt = blockIdx.x;   // 0..63
  const int bh = blockIdx.y;   // 0..31  (b*16 + h)
  const size_t head = (size_t)bh * SEQ * D_HEAD;
  const _Float16* Q = Qh + head;
  const _Float16* K = Kh + head;
  const _Float16* V = Vh + head;
  const int qbase = qt * 32;

  // Q fragments: 2 row sub-tiles x 2 K-steps (Dh = 64)
  v16h qa[2][2];
#pragma unroll
  for (int u = 0; u < 2; ++u) {
    const _Float16* qrow = Q + (size_t)(qbase + u * 16 + lm) * D_HEAD + 8 * hl;
    qa[u][0] = frag16(qrow);
    qa[u][1] = frag16(qrow + 32);
  }

  v8f o[2][4] = {};
  float rmax = -1e30f, rsum = 0.0f;  // every lane owns row qbase+lane

  const int kend = qbase + 32;  // causal upper bound (exclusive)
  for (int kv = 0; kv < kend; kv += 32) {
    // ---- kick off V block stage (32 x 64 f16), one row per lane
#if HAVE_ASYNC_LDS
    {
      GAS v4i* gsrc = (GAS v4i*)(V + (size_t)(kv + lane) * D_HEAD);
      LAS v4i* ldst = (LAS v4i*)(&sV[lane][0]);
#pragma unroll
      for (int i = 0; i < 8; ++i)
        __builtin_amdgcn_global_load_async_to_lds_b128(gsrc + i, ldst + i, 0, 0);
    }
#else
    {
      const float4* vr = (const float4*)(V + (size_t)(kv + lane) * D_HEAD);
      float4* dst = (float4*)(&sV[lane][0]);
#pragma unroll
      for (int i = 0; i < 8; ++i) dst[i] = vr[i];
    }
#endif
    // ---- scores for two 16-wide key tiles: S = (Q K^T) / 8, causal mask
#pragma unroll
    for (int t = 0; t < 2; ++t) {
      const _Float16* kr = K + (size_t)(kv + t * 16 + lm) * D_HEAD + 8 * hl;
      const v16h kb0 = frag16(kr);
      const v16h kb1 = frag16(kr + 32);
#pragma unroll
      for (int u = 0; u < 2; ++u) {
        v8f s = {};
        s = wmma16x16x32(qa[u][0], kb0, s);
        s = wmma16x16x32(qa[u][1], kb1, s);
#pragma unroll
        for (int r = 0; r < 8; ++r) {
          const int qg = qbase + u * 16 + r + 8 * hl;
          const int kg = kv + t * 16 + lm;
          float val = s[r] * 0.125f;          // 1/sqrt(64)
          if (kg > qg) val = -1e30f;          // causal mask (finite -> exp==0)
          sS[u * 16 + r + 8 * hl][t * 16 + lm] = val;
        }
      }
    }
#if HAVE_ASYNC_LDS
    __builtin_amdgcn_s_wait_asynccnt(0);
#endif
    __syncthreads();
    // ---- online softmax: every lane owns one q-row
    {
      float rowv[32];
      float bm = -1e30f;
#pragma unroll
      for (int j = 0; j < 32; ++j) {
        rowv[j] = sS[lane][j];
        bm = fmaxf(bm, rowv[j]);
      }
      const float mnew  = fmaxf(rmax, bm);
      const float alpha = __expf(rmax - mnew);
      float bsum = 0.0f;
#pragma unroll
      for (int j = 0; j < 32; ++j) {
        const float p = __expf(rowv[j] - mnew);
        bsum += p;
        sP[lane][j] = (_Float16)p;
      }
      rsum = rsum * alpha + bsum;
      rmax = mnew;
      sScale[lane] = alpha;
    }
    __syncthreads();
    // ---- rescale O accumulators by per-row alpha
#pragma unroll
    for (int u = 0; u < 2; ++u) {
      float al[8];
#pragma unroll
      for (int r = 0; r < 8; ++r) al[r] = sScale[u * 16 + r + 8 * hl];
#pragma unroll
      for (int t = 0; t < 4; ++t)
#pragma unroll
        for (int r = 0; r < 8; ++r) o[u][t][r] *= al[r];
    }
    // ---- O += P(32x32) @ Vblk(32x64): V fragment shared across q sub-tiles
    const v16h pa0 = frag16(&sP[lm][8 * hl]);
    const v16h pa1 = frag16(&sP[16 + lm][8 * hl]);
#pragma unroll
    for (int t = 0; t < 4; ++t) {
      const int d = t * 16 + lm;
      v16h vb;  // column-major gather of V block from LDS
#pragma unroll
      for (int e = 0; e < 8; ++e) {
        vb[e]     = sV[8 * hl + e][d];
        vb[e + 8] = sV[16 + 8 * hl + e][d];
      }
      o[0][t] = wmma16x16x32(pa0, vb, o[0][t]);
      o[1][t] = wmma16x16x32(pa1, vb, o[1][t]);
    }
    __syncthreads();
  }
  // ---- finalize: divide by row sums, store merged-head layout
  sSum[lane] = rsum;
  __syncthreads();
  const int b = bh >> 4, h = bh & 15;
#pragma unroll
  for (int u = 0; u < 2; ++u) {
    float inv[8];
#pragma unroll
    for (int r = 0; r < 8; ++r) inv[r] = 1.0f / sSum[u * 16 + r + 8 * hl];
#pragma unroll
    for (int t = 0; t < 4; ++t)
#pragma unroll
      for (int r = 0; r < 8; ++r) {
        const int qg = qbase + u * 16 + r + 8 * hl;
        Hm[((size_t)b * SEQ + qg) * D_MODEL + h * D_HEAD + t * 16 + lm] =
            (_Float16)(o[u][t][r] * inv[r]);
      }
  }
}

// ---------------------------------------------------------------------------
// Kernel 3: out[tok,n] (f32) = Hm(f16) @ Wo(f16)^T + bo, 32x64 wave tiles
// ---------------------------------------------------------------------------
__global__ __launch_bounds__(256)
void out_proj_kernel(const _Float16* __restrict__ Hm,
                     const _Float16* __restrict__ Wo,
                     const float* __restrict__ bo,
                     float* __restrict__ out) {
  const int lane = threadIdx.x & 31;
  const int wave = threadIdx.x >> 5;
  const int w = blockIdx.x * 8 + wave;   // 0..2047
  const int mtile = w >> 4;              // 0..127
  const int n64 = w & 15;
  const int hl = lane >> 4, lm = lane & 15;

  const _Float16* h0 = Hm + (size_t)(mtile * 32 + lm) * D_MODEL + 8 * hl;
  const _Float16* h1 = h0 + (size_t)16 * D_MODEL;
  const _Float16* wb = Wo + (size_t)(n64 * 64 + lm) * D_MODEL + 8 * hl;

  v8f acc[2][4] = {};
#pragma unroll 2
  for (int k0 = 0; k0 < D_MODEL; k0 += 32) {
    const v16h a0 = frag16(h0 + k0);
    const v16h a1 = frag16(h1 + k0);
    __builtin_prefetch(wb + k0 + 32, 0, 1);
#pragma unroll
    for (int t = 0; t < 4; ++t) {
      const v16h bf = frag16(wb + (size_t)t * 16 * D_MODEL + k0);
      acc[0][t] = wmma16x16x32(a0, bf, acc[0][t]);
      acc[1][t] = wmma16x16x32(a1, bf, acc[1][t]);
    }
  }
#pragma unroll
  for (int t = 0; t < 4; ++t) {
    const int ncol = n64 * 64 + t * 16 + lm;
    const float bb = bo[ncol];
#pragma unroll
    for (int u = 0; u < 2; ++u)
#pragma unroll
      for (int r = 0; r < 8; ++r) {
        const int tok = mtile * 32 + u * 16 + r + 8 * hl;
        out[(size_t)tok * D_MODEL + ncol] = acc[u][t][r] + bb;
      }
  }
}

// ---------------------------------------------------------------------------
extern "C" void kernel_launch(void* const* d_in, const int* in_sizes, int n_in,
                              void* d_out, int out_size, void* d_ws, size_t ws_size,
                              hipStream_t stream) {
  (void)in_sizes; (void)n_in; (void)out_size; (void)ws_size;
  const float* q  = (const float*)d_in[0];
  const float* k  = (const float*)d_in[1];
  const float* v  = (const float*)d_in[2];
  const float* Wq = (const float*)d_in[3];
  const float* bq = (const float*)d_in[4];
  const float* Wk = (const float*)d_in[5];
  const float* bk = (const float*)d_in[6];
  const float* Wv = (const float*)d_in[7];
  const float* bv = (const float*)d_in[8];
  const float* Wo = (const float*)d_in[9];
  const float* bo = (const float*)d_in[10];
  // d_in[11]: attn_mask = triu(ones,k=1) -> causal; applied analytically.

  const size_t MB = 1024 * 1024;
  char* ws = (char*)d_ws;
  _Float16* Qh   = (_Float16*)(ws + 0 * MB);   // 8 MB each
  _Float16* Kh   = (_Float16*)(ws + 8 * MB);
  _Float16* Vh   = (_Float16*)(ws + 16 * MB);
  _Float16* Hm   = (_Float16*)(ws + 24 * MB);
  _Float16* Xq16 = (_Float16*)(ws + 32 * MB);  // converted activations (8 MB)
  _Float16* Xk16 = (_Float16*)(ws + 40 * MB);
  _Float16* Xv16 = (_Float16*)(ws + 48 * MB);
  _Float16* Wq16 = (_Float16*)(ws + 56 * MB);  // converted weights (2 MB each)
  _Float16* Wk16 = (_Float16*)(ws + 58 * MB);
  _Float16* Wv16 = (_Float16*)(ws + 60 * MB);
  _Float16* Wo16 = (_Float16*)(ws + 62 * MB);

  // ---- one-time f32 -> f16 conversions (bandwidth-bound, ~96 MB total)
  const int actN8 = NTOK * D_MODEL / 8;       // 524288
  const int wN8   = D_MODEL * D_MODEL / 8;    // 131072
  dim3 bc(256);
  hipLaunchKernelGGL(cvt_f16_kernel, dim3((actN8 + 255) / 256), bc, 0, stream, q,  Xq16, actN8);
  hipLaunchKernelGGL(cvt_f16_kernel, dim3((actN8 + 255) / 256), bc, 0, stream, k,  Xk16, actN8);
  hipLaunchKernelGGL(cvt_f16_kernel, dim3((actN8 + 255) / 256), bc, 0, stream, v,  Xv16, actN8);
  hipLaunchKernelGGL(cvt_f16_kernel, dim3((wN8 + 255) / 256),   bc, 0, stream, Wq, Wq16, wN8);
  hipLaunchKernelGGL(cvt_f16_kernel, dim3((wN8 + 255) / 256),   bc, 0, stream, Wk, Wk16, wN8);
  hipLaunchKernelGGL(cvt_f16_kernel, dim3((wN8 + 255) / 256),   bc, 0, stream, Wv, Wv16, wN8);
  hipLaunchKernelGGL(cvt_f16_kernel, dim3((wN8 + 255) / 256),   bc, 0, stream, Wo, Wo16, wN8);

  // ---- QKV projections (pure-f16 WMMA GEMMs)
  dim3 gproj(256), bproj(256);
  hipLaunchKernelGGL(qkv_proj_kernel, gproj, bproj, 0, stream, Xq16, Wq16, bq, Qh);
  hipLaunchKernelGGL(qkv_proj_kernel, gproj, bproj, 0, stream, Xk16, Wk16, bk, Kh);
  hipLaunchKernelGGL(qkv_proj_kernel, gproj, bproj, 0, stream, Xv16, Wv16, bv, Vh);

  // ---- causal flash attention (32-row q tiles)
  dim3 gattn(SEQ / 32, BATCH * N_HEADS), battn(32);
  hipLaunchKernelGGL(flash_attn_kernel, gattn, battn, 0, stream, Qh, Kh, Vh, Hm);

  // ---- output projection
  hipLaunchKernelGGL(out_proj_kernel, gproj, bproj, 0, stream, Hm, Wo16, bo,
                     (float*)d_out);
}
